// GATModelWithAttention_807453852439
// MI455X (gfx1250) — matrix operand order repaired and checked by reading, not compiled, not run on hardware
//
#include <hip/hip_runtime.h>
#include <math.h>

// GAT model: N=50000 nodes, E=800000 edges (+N self loops), F=128, H=4, D=64,
// G=32 graphs, C=3 classes.
// Dense node GEMMs -> bf16 WMMA (v_wmma_f32_16x16x32_bf16, f32 accumulate),
// LDS tiles stored as packed bf16-pairs so fragment fetches are ds_load_b128.
// Edge softmax/aggregation -> float atomics (memory-bound irregular phase).

#define FDIM    128
#define HEADS   4
#define DHEAD   64
#define GRAPHS  32
#define CLASSES 3
#define HD      (HEADS * DHEAD)   // 256

typedef __attribute__((ext_vector_type(16))) __bf16        v16bf;
typedef __attribute__((ext_vector_type(8)))  float         v8f;
typedef __attribute__((ext_vector_type(4)))  unsigned int  uint4v;

union Frag { v16bf bf; uint4v u4[2]; };

__device__ __forceinline__ unsigned int pack_bf16x2(float a, float b) {
  unsigned short lo = __builtin_bit_cast(unsigned short, (__bf16)a);
  unsigned short hi = __builtin_bit_cast(unsigned short, (__bf16)b);
  return (unsigned int)lo | ((unsigned int)hi << 16);
}

// ---------------------------------------------------------------------------
// float atomic max via ordered-int trick (works for mixed signs, -inf init)
// ---------------------------------------------------------------------------
__device__ __forceinline__ void atomicMaxF(float* addr, float v) {
  if (v >= 0.0f) atomicMax((int*)addr, __float_as_int(v));
  else           atomicMin((unsigned int*)addr, __float_as_uint(v));
}

__device__ __forceinline__ int edge_end(const int* p, int e, int E) {
  return (e < E) ? p[e] : (e - E);   // synthesized self-loops for e >= E
}

// ---------------------------------------------------------------------------
// C[M,Nc] = A[M,K] @ B[K,Nc]; fp32 in/out, bf16 WMMA, f32 accumulate.
// Block: 128 threads = 4 waves. Block tile 128 rows x 64 cols.
// Wave w owns rows [w*32, w*32+32): 2 row-fragments x 4 col-tiles = 8 WMMAs
// per K-step; B fragments loaded once, reused for both row fragments.
//
// LDS layout: bf16 pairs packed into dwords, fragment-ordered:
//   Aspk[r][j]  = (A[r][k0+2j], A[r][k0+2j+1])           j = 0..15
//   Bpk [c][j]  = (B[k0+2j][c], B[k0+2j+1][c])           j = 0..15
// Row stride 20 dwords => every 4-dword fragment run is 16B-aligned
// => two ds_load_b128 per fragment.
//
// CDNA5 16-bit A fragment (16x32): lane m=lane&15, half=lane>>4;
//   pair p for VGPR r: r<4 -> p=r+4*half ; r>=4 -> p=8+(r-4)+4*half
//   => dwords [4*half .. +3] and [8+4*half .. +3].
// B fragment (32x16): lane n=lane&15; same pair order as A
//   => dwords [4*half .. +3] and [8+4*half .. +3].
// Requires: K % 32 == 0, Nc % 64 == 0 (true for all three layers).
// ---------------------------------------------------------------------------
__global__ void __launch_bounds__(128)
gemm_wmma_bf16(const float* __restrict__ A, const float* __restrict__ B,
               float* __restrict__ C, int M, int K, int Nc)
{
  __shared__ __attribute__((aligned(16))) unsigned int Aspk[128][20];
  __shared__ __attribute__((aligned(16))) unsigned int Bpk[64][20];

  const int tid  = threadIdx.x;
  const int wave = tid >> 5;
  const int lane = tid & 31;
  const int row0 = blockIdx.y * 128;
  const int col0 = blockIdx.x * 64;
  const int m    = lane & 15;
  const int half = lane >> 4;

  v8f acc[2][4] = {};

  for (int k0 = 0; k0 < K; k0 += 32) {
    // Stage A: 128 rows x 16 k-pairs (coalesced float2 loads)
    for (int i = tid; i < 128 * 16; i += 128) {
      int r = i >> 4, j = i & 15;
      int gr = row0 + r;
      float2 v = make_float2(0.f, 0.f);
      if (gr < M) v = *(const float2*)(A + (size_t)gr * K + k0 + 2 * j);
      Aspk[r][j] = pack_bf16x2(v.x, v.y);
    }
    // Stage B: 16 k-pairs x 64 cols (two coalesced row reads per pair)
    for (int i = tid; i < 16 * 64; i += 128) {
      int j = i >> 6, c = i & 63;
      float b0 = B[(size_t)(k0 + 2 * j) * Nc + col0 + c];
      float b1 = B[(size_t)(k0 + 2 * j + 1) * Nc + col0 + c];
      Bpk[c][j] = pack_bf16x2(b0, b1);
    }
    __syncthreads();

    // B fragments for the 4 column tiles (2x ds_load_b128 each)
    Frag bfr[4];
    #pragma unroll
    for (int t = 0; t < 4; ++t) {
      const unsigned int* bp = &Bpk[t * 16 + m][0];
      bfr[t].u4[0] = *(const uint4v*)(bp + 4 * half);
      bfr[t].u4[1] = *(const uint4v*)(bp + 8 + 4 * half);
    }

    #pragma unroll
    for (int rb = 0; rb < 2; ++rb) {
      Frag af;
      const unsigned int* ap = &Aspk[wave * 32 + rb * 16 + m][0];
      af.u4[0] = *(const uint4v*)(ap + 4 * half);
      af.u4[1] = *(const uint4v*)(ap + 8 + 4 * half);
      #pragma unroll
      for (int t = 0; t < 4; ++t) {
        acc[rb][t] = __builtin_amdgcn_wmma_f32_16x16x32_bf16(
            false, af.bf, false, bfr[t].bf, (short)0, acc[rb][t], false, false);
      }
    }
    __syncthreads();
  }

  // Epilogue. C/D layout: VGPR r -> M = r + 8*half, N = lane&15.
  // Fast path for full row-blocks (all but the last blockIdx.y): no
  // per-element predication, hoisted row pointers, immediate col offsets.
  const int n = lane & 15;
  const int rowbase = row0 + wave * 32 + half * 8;
  float* cbase = C + (size_t)rowbase * Nc + col0 + n;
  if (row0 + 128 <= M) {
    #pragma unroll
    for (int rb = 0; rb < 2; ++rb) {
      #pragma unroll
      for (int r = 0; r < 8; ++r) {
        float* crow = cbase + (size_t)(rb * 16 + r) * Nc;
        #pragma unroll
        for (int t = 0; t < 4; ++t) crow[t * 16] = acc[rb][t][r];
      }
    }
  } else {
    #pragma unroll
    for (int rb = 0; rb < 2; ++rb) {
      #pragma unroll
      for (int r = 0; r < 8; ++r) {
        if (rowbase + rb * 16 + r < M) {
          float* crow = cbase + (size_t)(rb * 16 + r) * Nc;
          #pragma unroll
          for (int t = 0; t < 4; ++t) crow[t * 16] = acc[rb][t][r];
        }
      }
    }
  }
}

// ---------------------------------------------------------------------------
// Per-node attention projections: al_s[n,h] = sum_d h[n,h,d]*a_src[h,d]
// ---------------------------------------------------------------------------
__global__ void compute_al(const float* __restrict__ h,
                           const float* __restrict__ a_s,
                           const float* __restrict__ a_d,
                           float* __restrict__ al_s, float* __restrict__ al_d,
                           int N, int Hl, int Dl)
{
  int idx = blockIdx.x * blockDim.x + threadIdx.x;
  if (idx >= N * Hl) return;
  int n = idx / Hl, hh = idx % Hl;
  const float* hp  = h + (size_t)n * Hl * Dl + hh * Dl;
  const float* asp = a_s + hh * Dl;
  const float* adp = a_d + hh * Dl;
  float ss = 0.f, sd = 0.f;
  for (int d = 0; d < Dl; ++d) { float v = hp[d]; ss += v * asp[d]; sd += v * adp[d]; }
  al_s[idx] = ss; al_d[idx] = sd;
}

__global__ void init_layer(float* agg, float* m, float* s, int nAgg, int nMS)
{
  int i = blockIdx.x * blockDim.x + threadIdx.x;
  if (i < nAgg) agg[i] = 0.f;
  if (i < nMS) { m[i] = -INFINITY; s[i] = 0.f; }
}

__global__ void edge_score(const int* src, const int* dst, int E, int Eall, int Hl,
                           const float* __restrict__ al_s,
                           const float* __restrict__ al_d,
                           float* __restrict__ ev, float* __restrict__ m)
{
  long i = (long)blockIdx.x * blockDim.x + threadIdx.x;
  if (i >= (long)Eall * Hl) return;
  int e = (int)(i / Hl), hh = (int)(i % Hl);
  int s_ = edge_end(src, e, E), d_ = edge_end(dst, e, E);
  float v = al_s[s_ * Hl + hh] + al_d[d_ * Hl + hh];
  v = (v > 0.f) ? v : 0.2f * v;   // leaky relu, slope 0.2
  ev[i] = v;
  atomicMaxF(&m[d_ * Hl + hh], v);
}

__global__ void edge_exp(const int* dst, int E, int Eall, int Hl,
                         const float* __restrict__ m,
                         float* __restrict__ ev, float* __restrict__ ssum)
{
  long i = (long)blockIdx.x * blockDim.x + threadIdx.x;
  if (i >= (long)Eall * Hl) return;
  int e = (int)(i / Hl), hh = (int)(i % Hl);
  int d_ = edge_end(dst, e, E);
  float ex = __expf(ev[i] - m[d_ * Hl + hh]);
  ev[i] = ex;
  atomicAdd(&ssum[d_ * Hl + hh], ex);
}

__global__ void edge_alpha(const int* dst, int E, int Eall, int Hl,
                           const float* __restrict__ ev,
                           const float* __restrict__ ssum,
                           float* __restrict__ att)
{
  long i = (long)blockIdx.x * blockDim.x + threadIdx.x;
  if (i >= (long)Eall * Hl) return;
  int e = (int)(i / Hl), hh = (int)(i % Hl);
  int d_ = edge_end(dst, e, E);
  att[i] = ev[i] / ssum[d_ * Hl + hh];
}

__global__ void edge_scatter(const int* src, const int* dst, int E, int Eall,
                             int Hl, int Dl,
                             const float* __restrict__ h,
                             const float* __restrict__ att,
                             float* __restrict__ agg)
{
  long i = (long)blockIdx.x * blockDim.x + threadIdx.x;
  int hd = Hl * Dl;
  if (i >= (long)Eall * hd) return;
  int e = (int)(i / hd), c = (int)(i % hd);
  int hh = c / Dl;
  int s_ = edge_end(src, e, E), d_ = edge_end(dst, e, E);
  float a = att[(size_t)e * Hl + hh];
  atomicAdd(&agg[(size_t)d_ * hd + c], h[(size_t)s_ * hd + c] * a);
}

__global__ void finalize_bias_act(float* __restrict__ agg,
                                  const float* __restrict__ b,
                                  int N, int hd, int do_elu)
{
  long i = (long)blockIdx.x * blockDim.x + threadIdx.x;
  if (i >= (long)N * hd) return;
  float v = agg[i] + b[i % hd];
  if (do_elu) v = (v > 0.f) ? v : (__expf(v) - 1.f);
  agg[i] = v;
}

__global__ void pool_init(float* pooled, float* counts)
{
  int i = blockIdx.x * blockDim.x + threadIdx.x;
  if (i < GRAPHS * DHEAD) pooled[i] = 0.f;
  if (i < GRAPHS) counts[i] = 0.f;
}

__global__ void pool_accum(const float* __restrict__ h3, const int* __restrict__ batch,
                           float* pooled, float* counts, int N)
{
  long i = (long)blockIdx.x * blockDim.x + threadIdx.x;
  if (i >= (long)N * DHEAD) return;
  int n = (int)(i / DHEAD), d = (int)(i % DHEAD);
  int g = batch[n];
  atomicAdd(&pooled[g * DHEAD + d], h3[i]);
  if (d == 0) atomicAdd(&counts[g], 1.f);
}

__global__ void classifier(const float* __restrict__ pooled,
                           const float* __restrict__ counts,
                           const float* __restrict__ Wc1, const float* __restrict__ bc1,
                           const float* __restrict__ Wc2, const float* __restrict__ bc2,
                           float* __restrict__ out)
{
  int g = threadIdx.x;
  if (g >= GRAPHS) return;
  float cnt = fmaxf(counts[g], 1.f);
  float pm[DHEAD];
  for (int d = 0; d < DHEAD; ++d) pm[d] = pooled[g * DHEAD + d] / cnt;
  float hid[DHEAD / 2];
  for (int j = 0; j < DHEAD / 2; ++j) {
    float v = bc1[j];
    for (int d = 0; d < DHEAD; ++d) v += pm[d] * Wc1[d * (DHEAD / 2) + j];
    hid[j] = fmaxf(v, 0.f);
  }
  for (int c = 0; c < CLASSES; ++c) {
    float v = bc2[c];
    for (int j = 0; j < DHEAD / 2; ++j) v += hid[j] * Wc2[j * CLASSES + c];
    out[g * CLASSES + c] = v;
  }
}

// ---------------------------------------------------------------------------
// One GAT layer: GEMM -> attention softmax -> scatter -> bias/ELU.
// Input features "in" [N,K]; transformed features -> bufA [N, Hl*64];
// aggregated + activated output -> bufB (safe: GEMM input dead after GEMM).
// ---------------------------------------------------------------------------
static void run_gat_layer(const float* in, int K,
                          const float* W, const float* a_s, const float* a_d,
                          const float* b, int Hl, int do_elu,
                          const int* src, const int* dst, int N, int E, int Eall,
                          float* bufA, float* bufB,
                          float* al_s, float* al_d, float* m, float* s,
                          float* ev, float* att, hipStream_t stream)
{
  const int Dl = DHEAD;
  const int Nc = Hl * Dl;

  dim3 ggrid((unsigned)(Nc / 64), (unsigned)((N + 127) / 128));
  gemm_wmma_bf16<<<ggrid, 128, 0, stream>>>(in, W, bufA, N, K, Nc);

  int nh = N * Hl;
  compute_al<<<(nh + 255) / 256, 256, 0, stream>>>(bufA, a_s, a_d, al_s, al_d, N, Hl, Dl);

  int nAgg = N * Nc;
  init_layer<<<(nAgg + 255) / 256, 256, 0, stream>>>(bufB, m, s, nAgg, nh);

  long eh = (long)Eall * Hl;
  unsigned ehb = (unsigned)((eh + 255) / 256);
  edge_score<<<ehb, 256, 0, stream>>>(src, dst, E, Eall, Hl, al_s, al_d, ev, m);
  edge_exp  <<<ehb, 256, 0, stream>>>(dst, E, Eall, Hl, m, ev, s);
  edge_alpha<<<ehb, 256, 0, stream>>>(dst, E, Eall, Hl, ev, s, att);

  long tot = (long)Eall * Nc;
  edge_scatter<<<(unsigned)((tot + 255) / 256), 256, 0, stream>>>(
      src, dst, E, Eall, Hl, Dl, bufA, att, bufB);

  finalize_bias_act<<<(unsigned)(((long)nAgg + 255) / 256), 256, 0, stream>>>(
      bufB, b, N, Nc, do_elu);
}

extern "C" void kernel_launch(void* const* d_in, const int* in_sizes, int n_in,
                              void* d_out, int out_size, void* d_ws, size_t ws_size,
                              hipStream_t stream)
{
  const float* x          = (const float*)d_in[0];
  const int*   edge_index = (const int*)d_in[1];   // harness delivers ints
  const int*   batch      = (const int*)d_in[2];
  const float* W1  = (const float*)d_in[3];
  const float* as1 = (const float*)d_in[4];
  const float* ad1 = (const float*)d_in[5];
  const float* b1  = (const float*)d_in[6];
  const float* W2  = (const float*)d_in[7];
  const float* as2 = (const float*)d_in[8];
  const float* ad2 = (const float*)d_in[9];
  const float* b2  = (const float*)d_in[10];
  const float* W3  = (const float*)d_in[11];
  const float* as3 = (const float*)d_in[12];
  const float* ad3 = (const float*)d_in[13];
  const float* b3  = (const float*)d_in[14];
  const float* Wc1 = (const float*)d_in[15];
  const float* bc1 = (const float*)d_in[16];
  const float* Wc2 = (const float*)d_in[17];
  const float* bc2 = (const float*)d_in[18];

  const int N    = in_sizes[0] / FDIM;
  const int E    = in_sizes[1] / 2;
  const int Eall = E + N;           // add_self_loops
  const int* src = edge_index;
  const int* dst = edge_index + E;

  // workspace carve-up
  float* w = (float*)d_ws;
  float* bufA   = w; w += (size_t)N * HD;
  float* bufB   = w; w += (size_t)N * HD;
  float* al_s   = w; w += (size_t)N * HEADS;
  float* al_d   = w; w += (size_t)N * HEADS;
  float* mbuf   = w; w += (size_t)N * HEADS;
  float* sbuf   = w; w += (size_t)N * HEADS;
  float* ev     = w; w += (size_t)Eall * HEADS;
  float* pooled = w; w += GRAPHS * DHEAD;
  float* counts = w; w += GRAPHS;

  // output carve-up: [G*C] out, then att1 [Eall*H], att2 [Eall*H], att3 [Eall]
  float* out  = (float*)d_out;
  float* att1 = out + GRAPHS * CLASSES;
  float* att2 = att1 + (size_t)Eall * HEADS;
  float* att3 = att2 + (size_t)Eall * HEADS;

  // layer 1: x[N,128] -> h1[N,256] (bufB)
  run_gat_layer(x, FDIM, W1, as1, ad1, b1, HEADS, 1,
                src, dst, N, E, Eall, bufA, bufB,
                al_s, al_d, mbuf, sbuf, ev, att1, stream);
  // layer 2: h1[N,256] -> h2[N,256] (bufB)
  run_gat_layer(bufB, HD, W2, as2, ad2, b2, HEADS, 1,
                src, dst, N, E, Eall, bufA, bufB,
                al_s, al_d, mbuf, sbuf, ev, att2, stream);
  // layer 3: h2[N,256] -> h3[N,64] (bufB), single head, no ELU, no concat
  run_gat_layer(bufB, HD, W3, as3, ad3, b3, 1, 0,
                src, dst, N, E, Eall, bufA, bufB,
                al_s, al_d, mbuf, sbuf, ev, att3, stream);

  // global mean pool + classifier
  pool_init<<<(GRAPHS * DHEAD + 255) / 256, 256, 0, stream>>>(pooled, counts);
  pool_accum<<<(unsigned)(((long)N * DHEAD + 255) / 256), 256, 0, stream>>>(
      bufB, batch, pooled, counts, N);
  classifier<<<1, 32, 0, stream>>>(pooled, counts, Wc1, bc1, Wc2, bc2, out);
}
